// DecoderMinLSTMGNN_24962349924442
// MI455X (gfx1250) — matrix operand order repaired and checked by reading, not compiled, not run on hardware
//
#include <hip/hip_runtime.h>

// ---------------------------------------------------------------------------
// MinLSTM (2 layers) + residual + LayerNorm + D->1 projection for MI455X
// (gfx1250, wave32, WMMA).  f32 in/out, bf16 WMMA math with f32 accumulate.
// ---------------------------------------------------------------------------

#define D_DIM  512
#define T_DIM  4096
#define B_DIM  8
#define M_DIM  (B_DIM * T_DIM)     // 32768 tokens
#define CHUNK  128                 // scan chunk length
#define NCH    (T_DIM / CHUNK)     // 32 chunks per sequence
#define LN_EPSF 1e-5f

typedef __attribute__((ext_vector_type(16))) __bf16 bf16x16;
typedef __attribute__((ext_vector_type(8)))  float  f32x8;

union FragBF { bf16x16 v; unsigned u[8]; };

// f32 pair -> packed bf16x2 (round-to-nearest-even, inf/nan not special-cased)
__device__ __forceinline__ unsigned pack_bf16(float a, float b) {
  unsigned ua = __float_as_uint(a);
  unsigned ub = __float_as_uint(b);
  ua += 0x7FFFu + ((ua >> 16) & 1u);
  ub += 0x7FFFu + ((ub >> 16) & 1u);
  return (ua >> 16) | (ub & 0xFFFF0000u);
}

__device__ __forceinline__ float sigmoidf_fast(float x) {
  return 1.0f / (1.0f + __expf(-x));
}

// ---------------------------------------------------------------------------
// Fused triple GEMM (f, i, h~ pre-activations share the X tile) + minLSTM gate
// epilogue.  X:[M,512] f32, W*:[512,512] f32 (row = output feature, so
// C = X @ W^T means B-matrix column n == W row n).  Writes a = f/(f+i) and
// u = (i/(f+i))*h~ straight to HBM -- pre-activations never leave registers.
//
// Block: 256 threads = 8 waves.  Tile: BM=128, BN=64, BK=32.
// Wave w owns rows [16w,16w+16) x all 64 cols x 3 matrices = 12 accumulators.
// LDS rows padded to 17 dwords (17 coprime 64 banks -> conflict-free frags).
// ---------------------------------------------------------------------------
__global__ __launch_bounds__(256) void minlstm_gemm_gates(
    const float* __restrict__ X,
    const float* __restrict__ Wf, const float* __restrict__ bf,
    const float* __restrict__ Wi, const float* __restrict__ bi,
    const float* __restrict__ Wh, const float* __restrict__ bh,
    float* __restrict__ AG, float* __restrict__ UG)
{
  __shared__ unsigned sX[128 * 17];      // 128 rows x 32 bf16 (16 dw) + pad
  __shared__ unsigned sW[3][64 * 17];    // 3 mats x 64 rows x 32 bf16 + pad

  const int tid  = threadIdx.x;
  const int lane = tid & 31;
  const int w    = tid >> 5;             // wave 0..7
  const int bm   = blockIdx.y;           // 0..255  (M/128)
  const int bn   = blockIdx.x;           // 0..7    (N/64)
  const int kh   = lane >> 4;            // lane half (ISA K split)

  const float* Wmat[3] = {Wf, Wi, Wh};

  f32x8 acc[3][4];
  #pragma unroll
  for (int m = 0; m < 3; ++m)
    #pragma unroll
    for (int nt = 0; nt < 4; ++nt)
      #pragma unroll
      for (int e = 0; e < 8; ++e) acc[m][nt][e] = 0.0f;

  for (int kk = 0; kk < 16; ++kk) {      // K = 512, BK = 32
    const int kbase = kk * 32;

    // ---- stage X tile 128x32 (f32 -> bf16 pairs), 8 dwords per thread
    #pragma unroll
    for (int j = 0; j < 8; ++j) {
      const int i  = tid + 256 * j;
      const int r  = i >> 4, c2 = i & 15;
      const float2 xv = *(const float2*)(X + (size_t)(bm * 128 + r) * D_DIM
                                           + kbase + 2 * c2);
      sX[r * 17 + c2] = pack_bf16(xv.x, xv.y);
    }
    // ---- stage the 3 weight tiles 64x32 each, 4 dwords per thread per mat
    #pragma unroll
    for (int m = 0; m < 3; ++m) {
      #pragma unroll
      for (int j = 0; j < 4; ++j) {
        const int i = tid + 256 * j;
        const int r = i >> 4, c2 = i & 15;
        const float2 wv = *(const float2*)(Wmat[m] + (size_t)(bn * 64 + r) * D_DIM
                                                   + kbase + 2 * c2);
        sW[m][r * 17 + c2] = pack_bf16(wv.x, wv.y);
      }
    }
    __syncthreads();

    // ---- A fragment (16x32 MxK): VGPR j<4 -> K=8*kh+2j ; j>=4 -> 16+8*kh+2(j-4)
    FragBF afrag;
    const int arow = w * 16 + (lane & 15);
    #pragma unroll
    for (int j = 0; j < 8; ++j) {
      const int k2 = (j < 4) ? (4 * kh + j) : (8 + 4 * kh + (j - 4));
      afrag.u[j] = sX[arow * 17 + k2];
    }

    // ---- B fragments (32x16 KxN): VGPR j<4 -> K=8*kh+2j ; j>=4 -> 16+8*kh+2(j-4)
    #pragma unroll
    for (int m = 0; m < 3; ++m) {
      #pragma unroll
      for (int nt = 0; nt < 4; ++nt) {
        FragBF bfrag;
        const int nrow = nt * 16 + (lane & 15);
        #pragma unroll
        for (int j = 0; j < 8; ++j) {
          const int k2 = (j < 4) ? (4 * kh + j) : (8 + 4 * kh + (j - 4));
          bfrag.u[j] = sW[m][nrow * 17 + k2];
        }
        acc[m][nt] = __builtin_amdgcn_wmma_f32_16x16x32_bf16(
            false, afrag.v, false, bfrag.v, (short)0, acc[m][nt], false, false);
      }
    }
    __syncthreads();
  }

  // ---- gate epilogue: C/D layout -> lane: N = lane&15, M = 8*(lane>>4)+r
  const int row_base = bm * 128 + w * 16 + ((lane >> 4) << 3);
  #pragma unroll
  for (int nt = 0; nt < 4; ++nt) {
    const int col = bn * 64 + nt * 16 + (lane & 15);
    const float bfv = bf[col], biv = bi[col], bhv = bh[col];
    #pragma unroll
    for (int r = 0; r < 8; ++r) {
      const int row = row_base + r;
      const float f  = sigmoidf_fast(acc[0][nt][r] + bfv);
      const float ii = sigmoidf_fast(acc[1][nt][r] + biv);
      const float hp = acc[2][nt][r] + bhv;
      const float dn = f + ii;
      AG[(size_t)row * D_DIM + col] = f / dn;
      UG[(size_t)row * D_DIM + col] = (ii / dn) * hp;
    }
  }
}

// ---------------------------------------------------------------------------
// 3-pass chunked linear-recurrence scan over T:  h_t = a_t*h_{t-1} + u_t
// Pass 1: per (b, chunk, d) local scan -> composed op (prod a, local h)
// ---------------------------------------------------------------------------
__global__ __launch_bounds__(256) void scan_chunks(
    const float* __restrict__ AG, const float* __restrict__ UG,
    float* __restrict__ cA, float* __restrict__ cU)
{
  const int tid = blockIdx.x * 256 + threadIdx.x;   // [b][ch][d] linear
  const int d  = tid & (D_DIM - 1);
  const int ch = (tid >> 9) & (NCH - 1);
  const int b  = tid >> 14;
  const size_t base = (size_t)(b * T_DIM + ch * CHUNK) * D_DIM + d;
  float ap = 1.0f, h = 0.0f;
  for (int t = 0; t < CHUNK; ++t) {
    const float a = AG[base + (size_t)t * D_DIM];
    const float u = UG[base + (size_t)t * D_DIM];
    ap *= a;
    h = a * h + u;
  }
  cA[tid] = ap;
  cU[tid] = h;
}

// Pass 2: per (b,d), sequential scan over the 32 chunk operators; rewrite cU
// in place with the carry-in h entering each chunk.
__global__ __launch_bounds__(256) void scan_carry(
    const float* __restrict__ cA, float* __restrict__ cU)
{
  const int tid = blockIdx.x * 256 + threadIdx.x;   // B*D threads
  const int d = tid & (D_DIM - 1);
  const int b = tid >> 9;
  float h = 0.0f;
  for (int ch = 0; ch < NCH; ++ch) {
    const int idx = (b * NCH + ch) * D_DIM + d;
    const float ap = cA[idx];
    const float hl = cU[idx];
    cU[idx] = h;                 // carry-in for this chunk
    h = ap * h + hl;
  }
}

// Pass 3: rescan each chunk seeded with its carry-in, write h out.
__global__ __launch_bounds__(256) void scan_apply(
    const float* __restrict__ AG, const float* __restrict__ UG,
    const float* __restrict__ cU, float* __restrict__ H)
{
  const int tid = blockIdx.x * 256 + threadIdx.x;
  const int d  = tid & (D_DIM - 1);
  const int ch = (tid >> 9) & (NCH - 1);
  const int b  = tid >> 14;
  const size_t base = (size_t)(b * T_DIM + ch * CHUNK) * D_DIM + d;
  float h = cU[tid];
  for (int t = 0; t < CHUNK; ++t) {
    const float a = AG[base + (size_t)t * D_DIM];
    const float u = UG[base + (size_t)t * D_DIM];
    h = a * h + u;
    H[base + (size_t)t * D_DIM] = h;
  }
}

// ---------------------------------------------------------------------------
// NaN-fallback + residual + LayerNorm + D->1 projection.
// One wave32 per token; 16 elements per lane; __shfl_xor reductions.
// ---------------------------------------------------------------------------
__global__ __launch_bounds__(256) void ln_proj(
    const float* __restrict__ X, const float* __restrict__ H2,
    const float* __restrict__ g, const float* __restrict__ be,
    const float* __restrict__ Wout, const float* __restrict__ bout,
    float* __restrict__ out)
{
  const int lane = threadIdx.x & 31;
  const int tok  = blockIdx.x * 8 + (threadIdx.x >> 5);
  const float* xp = X  + (size_t)tok * D_DIM;
  const float* hp = H2 + (size_t)tok * D_DIM;

  float v[16];
  float s = 0.0f;
  #pragma unroll
  for (int j = 0; j < 16; ++j) {
    const int d = lane + 32 * j;
    float hv = hp[d];
    const float xv = xp[d];
    if (hv != hv) hv = xv;       // NaN fallback to input
    hv += xv;                    // residual
    v[j] = hv;
    s += hv;
  }
  #pragma unroll
  for (int off = 16; off >= 1; off >>= 1) s += __shfl_xor(s, off, 32);
  const float mu = s * (1.0f / D_DIM);

  float sq = 0.0f;
  #pragma unroll
  for (int j = 0; j < 16; ++j) { const float t = v[j] - mu; sq += t * t; }
  #pragma unroll
  for (int off = 16; off >= 1; off >>= 1) sq += __shfl_xor(sq, off, 32);
  const float rstd = rsqrtf(sq * (1.0f / D_DIM) + LN_EPSF);

  float accd = 0.0f;
  #pragma unroll
  for (int j = 0; j < 16; ++j) {
    const int d = lane + 32 * j;
    const float hn = (v[j] - mu) * rstd * g[d] + be[d];
    accd += hn * Wout[d];
  }
  #pragma unroll
  for (int off = 16; off >= 1; off >>= 1) accd += __shfl_xor(accd, off, 32);
  if (lane == 0) out[tok] = accd + bout[0];
}

// ---------------------------------------------------------------------------
extern "C" void kernel_launch(void* const* d_in, const int* in_sizes, int n_in,
                              void* d_out, int out_size, void* d_ws, size_t ws_size,
                              hipStream_t stream) {
  (void)in_sizes; (void)n_in; (void)out_size; (void)ws_size;
  const float* x    = (const float*)d_in[0];
  const float* Wf0  = (const float*)d_in[1];
  const float* bf0  = (const float*)d_in[2];
  const float* Wi0  = (const float*)d_in[3];
  const float* bi0  = (const float*)d_in[4];
  const float* Wh0  = (const float*)d_in[5];
  const float* bh0  = (const float*)d_in[6];
  const float* Wf1  = (const float*)d_in[7];
  const float* bf1  = (const float*)d_in[8];
  const float* Wi1  = (const float*)d_in[9];
  const float* bi1  = (const float*)d_in[10];
  const float* Wh1  = (const float*)d_in[11];
  const float* bh1  = (const float*)d_in[12];
  const float* ln_g = (const float*)d_in[13];
  const float* ln_b = (const float*)d_in[14];
  const float* Wout = (const float*)d_in[15];
  const float* bout = (const float*)d_in[16];
  float* out = (float*)d_out;

  // workspace: 3 x 64MB ping-pong buffers + carries (~203 MB total)
  float* AG = (float*)d_ws;
  float* UG = AG + (size_t)M_DIM * D_DIM;
  float* H  = UG + (size_t)M_DIM * D_DIM;
  float* cA = H  + (size_t)M_DIM * D_DIM;
  float* cU = cA + (size_t)B_DIM * D_DIM * NCH;

  const dim3 gg(8, 256);     // (N/64, M/128) tiles
  const dim3 gb(256);

  // ---- layer 0
  minlstm_gemm_gates<<<gg, gb, 0, stream>>>(x, Wf0, bf0, Wi0, bi0, Wh0, bh0, AG, UG);
  scan_chunks<<<512, 256, 0, stream>>>(AG, UG, cA, cU);
  scan_carry <<< 16, 256, 0, stream>>>(cA, cU);
  scan_apply <<<512, 256, 0, stream>>>(AG, UG, cU, H);
  // ---- layer 1 (reads H, AG/UG recycled)
  minlstm_gemm_gates<<<gg, gb, 0, stream>>>(H, Wf1, bf1, Wi1, bi1, Wh1, bh1, AG, UG);
  scan_chunks<<<512, 256, 0, stream>>>(AG, UG, cA, cU);
  scan_carry <<< 16, 256, 0, stream>>>(cA, cU);
  scan_apply <<<512, 256, 0, stream>>>(AG, UG, cU, H);   // H := h2
  // ---- residual + LN + output projection
  ln_proj<<<4096, 256, 0, stream>>>(x, H, ln_g, ln_b, Wout, bout, out);
}
